// NodeModel_53455162966482
// MI455X (gfx1250) — matrix operand order repaired and checked by reading, not compiled, not run on hardware
//
#include <hip/hip_runtime.h>

#define NNODES 10000
#define NEDGES 640000
#define HDIM   128
#define LNEPS  1e-5f
#define WPB    4        // waves per block in the MLP kernel
#define SSTRIDE 136     // f16 LDS row stride (272 B, 16B-aligned)

typedef __attribute__((ext_vector_type(16))) _Float16 v16h;
typedef __attribute__((ext_vector_type(8)))  _Float16 v8h;
typedef __attribute__((ext_vector_type(8)))  float    v8f;

// ---------------------------------------------------------------------------
// Repack W0 (256x128), W1, W2 (128x128) f32 row-major into f16 B-fragments.
// Fragment layout (per CDNA5 ISA 7.12.2, 16-bit B 32x16):
//   lane = n(0..15) + 16*khalf ; 16 f16 per lane = K (j<8 ? khalf*8 + j : 16 + khalf*8 + j-8).
// Fragment order: L0 frags 0..63 (ktile*8+ntile), L1 64..95, L2 96..127.
// Each lane's 16 f16 stored contiguously (32 B) -> coalesced b128 loads later.
// ---------------------------------------------------------------------------
__global__ void pack_weights_kernel(const float* __restrict__ W0,
                                    const float* __restrict__ W1,
                                    const float* __restrict__ W2,
                                    _Float16* __restrict__ wf) {
  int tid = blockIdx.x * blockDim.x + threadIdx.x;
  if (tid >= 128 * 32) return;
  int fragid = tid >> 5;
  int lane   = tid & 31;
  const float* W; int f;
  if (fragid < 64)      { W = W0; f = fragid; }
  else if (fragid < 96) { W = W1; f = fragid - 64; }
  else                  { W = W2; f = fragid - 96; }
  int ktile = f >> 3, ntile = f & 7;
  int n     = ntile * 16 + (lane & 15);
  int khalf = lane >> 4;
  _Float16* dst = wf + (size_t)tid * 16;
  #pragma unroll
  for (int j = 0; j < 16; ++j) {
    int k = ktile * 32 + (j < 8 ? khalf * 8 + j : 16 + khalf * 8 + (j - 8));
    dst[j] = (_Float16)W[k * HDIM + n];
  }
}

// ---------------------------------------------------------------------------
// Scatter-add edge features into receiver nodes. One thread per (edge, 4 cols).
// ---------------------------------------------------------------------------
__global__ void scatter_add_kernel(const float* __restrict__ e,
                                   const long long* __restrict__ ei,
                                   float* __restrict__ agg) {
  int t = blockIdx.x * blockDim.x + threadIdx.x;
  if (t >= NEDGES * (HDIM / 4)) return;
  int eidx = t >> 5;        // / (HDIM/4)
  int c4   = t & 31;
  int recv = (int)ei[(size_t)NEDGES + eidx];   // edge_index[1][eidx]
  const float4 val = *(const float4*)(e + (size_t)eidx * HDIM + c4 * 4);
  float* dst = agg + (size_t)recv * HDIM + c4 * 4;
  atomicAdd(dst + 0, val.x);
  atomicAdd(dst + 1, val.y);
  atomicAdd(dst + 2, val.z);
  atomicAdd(dst + 3, val.w);
}

// ---------------------------------------------------------------------------
// Hidden layer: acc[0..7] = (stage 16x128 f16) x (W fragments) , K = 128.
// srow points at stage[wid][m][0] (A-fragment rows per lane).
// ---------------------------------------------------------------------------
__device__ __forceinline__ void hidden_matmul(v8f acc[8],
                                              const _Float16* __restrict__ wf,
                                              int fragbase,
                                              const _Float16* srow,
                                              int g, int lane) {
  const v8f vzero = {};
  #pragma unroll
  for (int t = 0; t < 8; ++t) acc[t] = vzero;
  #pragma unroll
  for (int ks = 0; ks < 4; ++ks) {
    const v8h lo = *(const v8h*)(srow + ks * 32 + g * 8);
    const v8h hi = *(const v8h*)(srow + ks * 32 + g * 8 + 16);
    v16h a;
    #pragma unroll
    for (int i = 0; i < 8; ++i) { a[i] = lo[i]; a[8 + i] = hi[i]; }
    #pragma unroll
    for (int t = 0; t < 8; ++t) {
      const v16h b = *(const v16h*)(wf +
          (size_t)((fragbase + ks * 8 + t) * 32 + lane) * 16);
      acc[t] = __builtin_amdgcn_wmma_f32_16x16x32_f16(false, a, false, b,
                                                      (short)0, acc[t],
                                                      false, false);
    }
  }
}

// ---------------------------------------------------------------------------
// Fused MLP (3 layers, ReLU) + LayerNorm. One wave = 16 nodes.
// ---------------------------------------------------------------------------
__global__ __launch_bounds__(WPB * 32)
void mlp_ln_kernel(const float* __restrict__ agg,
                   const float* __restrict__ vin,
                   const _Float16* __restrict__ wf,
                   const float* __restrict__ b0, const float* __restrict__ b1,
                   const float* __restrict__ b2,
                   const float* __restrict__ ln_g,
                   const float* __restrict__ ln_b,
                   float* __restrict__ out) {
  __shared__ _Float16 stage[WPB][16][SSTRIDE];
  const int lane = threadIdx.x & 31;
  const int wid  = threadIdx.x >> 5;
  const int node_base = (blockIdx.x * WPB + wid) * 16;
  if (node_base >= NNODES) return;           // wave-uniform: EXEC stays full
  const int m = lane & 15;                   // matrix row within tile
  const int g = lane >> 4;                   // K-half group
  const size_t row = (size_t)(node_base + m);

  const v8f vzero = {};
  v8f acc[8];
  #pragma unroll
  for (int t = 0; t < 8; ++t) acc[t] = vzero;

  // ---- Layer 0: [16 x 256] @ W0[256 x 128], x = concat(agg, v) -------------
  #pragma unroll
  for (int ks = 0; ks < 8; ++ks) {
    const float* src = (ks < 4) ? (agg + row * HDIM + ks * 32)
                                : (vin + row * HDIM + (ks - 4) * 32);
    const float4 x0 = *(const float4*)(src + g * 8 + 0);
    const float4 x1 = *(const float4*)(src + g * 8 + 4);
    const float4 x2 = *(const float4*)(src + g * 8 + 16);
    const float4 x3 = *(const float4*)(src + g * 8 + 20);
    v16h a;
    a[0] = x0.x;  a[1] = x0.y;  a[2]  = x0.z;  a[3]  = x0.w;
    a[4] = x1.x;  a[5] = x1.y;  a[6]  = x1.z;  a[7]  = x1.w;
    a[8] = x2.x;  a[9] = x2.y;  a[10] = x2.z;  a[11] = x2.w;
    a[12] = x3.x; a[13] = x3.y; a[14] = x3.z;  a[15] = x3.w;
    #pragma unroll
    for (int t = 0; t < 8; ++t) {
      const v16h b = *(const v16h*)(wf + (size_t)((ks * 8 + t) * 32 + lane) * 16);
      acc[t] = __builtin_amdgcn_wmma_f32_16x16x32_f16(false, a, false, b,
                                                      (short)0, acc[t],
                                                      false, false);
    }
  }
  // bias + ReLU -> LDS stage (f16), C layout: row r+8g, col t*16+m
  #pragma unroll
  for (int t = 0; t < 8; ++t) {
    const float bias = b0[t * 16 + m];
    #pragma unroll
    for (int r = 0; r < 8; ++r) {
      float val = acc[t][r] + bias;
      stage[wid][r + 8 * g][t * 16 + m] = (_Float16)(val > 0.f ? val : 0.f);
    }
  }

  // ---- Layer 1 -------------------------------------------------------------
  hidden_matmul(acc, wf, 64, &stage[wid][m][0], g, lane);
  #pragma unroll
  for (int t = 0; t < 8; ++t) {
    const float bias = b1[t * 16 + m];
    #pragma unroll
    for (int r = 0; r < 8; ++r) {
      float val = acc[t][r] + bias;
      stage[wid][r + 8 * g][t * 16 + m] = (_Float16)(val > 0.f ? val : 0.f);
    }
  }

  // ---- Layer 2 (keep in registers) ----------------------------------------
  hidden_matmul(acc, wf, 96, &stage[wid][m][0], g, lane);
  #pragma unroll
  for (int t = 0; t < 8; ++t) {
    const float bias = b2[t * 16 + m];
    #pragma unroll
    for (int r = 0; r < 8; ++r) {
      float val = acc[t][r] + bias;
      acc[t][r] = val > 0.f ? val : 0.f;
    }
  }

  // ---- LayerNorm over H=128 per row ---------------------------------------
  float mu[8], rstd[8];
  #pragma unroll
  for (int r = 0; r < 8; ++r) {
    float s = 0.f, sq = 0.f;
    #pragma unroll
    for (int t = 0; t < 8; ++t) { const float x = acc[t][r]; s += x; sq += x * x; }
    #pragma unroll
    for (int off = 1; off < 16; off <<= 1) {   // reduce over the 16 n-lanes
      s  += __shfl_xor(s, off, 32);
      sq += __shfl_xor(sq, off, 32);
    }
    const float mean = s * (1.f / HDIM);
    const float var  = sq * (1.f / HDIM) - mean * mean;
    mu[r]   = mean;
    rstd[r] = rsqrtf(var + LNEPS);
  }
  #pragma unroll
  for (int t = 0; t < 8; ++t) {
    const float gg = ln_g[t * 16 + m];
    const float bb = ln_b[t * 16 + m];
    #pragma unroll
    for (int r = 0; r < 8; ++r) {
      const float y = (acc[t][r] - mu[r]) * rstd[r] * gg + bb;
      out[(size_t)(node_base + r + 8 * g) * HDIM + t * 16 + m] = y;
    }
  }
}

// ---------------------------------------------------------------------------
extern "C" void kernel_launch(void* const* d_in, const int* in_sizes, int n_in,
                              void* d_out, int out_size, void* d_ws, size_t ws_size,
                              hipStream_t stream) {
  const float*     v    = (const float*)d_in[0];
  const float*     e    = (const float*)d_in[1];
  const long long* ei   = (const long long*)d_in[2];
  const float*     W0   = (const float*)d_in[3];
  const float*     b0   = (const float*)d_in[4];
  const float*     W1   = (const float*)d_in[5];
  const float*     b1   = (const float*)d_in[6];
  const float*     W2   = (const float*)d_in[7];
  const float*     b2   = (const float*)d_in[8];
  const float*     ln_g = (const float*)d_in[9];
  const float*     ln_b = (const float*)d_in[10];
  float*           out  = (float*)d_out;

  const size_t agg_bytes = (size_t)NNODES * HDIM * sizeof(float);  // 5.12 MB
  float*     agg = (float*)d_ws;
  _Float16*  wf  = (_Float16*)((char*)d_ws + agg_bytes);           // 128 KB, 32B-aligned

  hipMemsetAsync(agg, 0, agg_bytes, stream);

  pack_weights_kernel<<<(128 * 32 + 255) / 256, 256, 0, stream>>>(W0, W1, W2, wf);

  const int sc_threads = NEDGES * (HDIM / 4);
  scatter_add_kernel<<<(sc_threads + 255) / 256, 256, 0, stream>>>(e, ei, agg);

  const int nwaves = (NNODES + 15) / 16;           // 625
  const int blocks = (nwaves + WPB - 1) / WPB;     // 157
  mlp_ln_kernel<<<blocks, WPB * 32, 0, stream>>>(agg, v, wf, b0, b1, b2,
                                                 ln_g, ln_b, out);
}
